// MultiValueHead_16458314678652
// MI455X (gfx1250) — matrix operand order repaired and checked by reading, not compiled, not run on hardware
//
#include <hip/hip_runtime.h>
#include <hip/hip_bf16.h>

typedef __attribute__((ext_vector_type(16))) __bf16 v16bf;
typedef __attribute__((ext_vector_type(8)))  float  v8f;
typedef __attribute__((ext_vector_type(4)))  float  v4f;

#define BATCH_SZ   32768
#define IN_DIM     2048
#define NCOLS      280     // 28 heads * 10 classes
#define NPAD       288     // 18 tiles of 16
#define NT         18
#define BM         128     // M rows per block
#define KC         32      // K per WMMA step
#define LDB        40      // bf16 elems per LDS column (80 B stride -> conflict-free)

__device__ __forceinline__ unsigned short f2bf(float f) {
    union { __bf16 h; unsigned short u; } cv;
    cv.h = (__bf16)f;
    return cv.u;
}

// ---------------- Kernel 0: W f32 -> bf16 ----------------
__global__ __launch_bounds__(256)
void wcvt_kernel(const float* __restrict__ W, unsigned short* __restrict__ Wb, int n) {
    int i = blockIdx.x * 256 + threadIdx.x;
    if (i < n) Wb[i] = f2bf(W[i]);
}

// ---- async DMA: global -> LDS, 16 B per lane, tracked by ASYNCcnt ----
__device__ __forceinline__ void async_b128(uint32_t lds_off, const void* gaddr) {
    asm volatile("global_load_async_to_lds_b128 %0, %1, off"
                 :: "v"(lds_off), "v"(gaddr) : "memory");
}
__device__ __forceinline__ void wait_async0() {
    asm volatile("s_wait_asynccnt 0" ::: "memory");
}

// stage one 288x32 bf16 B tile (layout [col][k], 80 B column stride) via async DMA
__device__ __forceinline__ void stageB_async(const unsigned short* Wb, int kc,
                                             uint32_t ldsBase, int tid) {
    const char* wb = (const char*)Wb + (size_t)kc * 2;   // bf16 bytes
#pragma unroll
    for (int i = 0; i < 5; ++i) {                        // 1152 groups of 16 B
        int g = i * 256 + tid;
        if (g < NPAD * 4) {
            int col  = g >> 2;
            int kq   = (g & 3) * 16;                     // bytes within column
            int scol = (col < NCOLS) ? col : (NCOLS - 1);   // clamp pad cols (no OOB)
            uint32_t loff   = ldsBase + (uint32_t)(col * (LDB * 2) + kq);
            const char* src = wb + (size_t)scol * (IN_DIM * 2) + kq;
            async_b128(loff, src);
        }
    }
}

// ---------------- Kernel 1: bf16 WMMA GEMM + bias ----------------
__global__ __launch_bounds__(256)
void gemm_bf16_kernel(const float* __restrict__ hidden,
                      const unsigned short* __restrict__ Wb,   // [280][2048] bf16
                      const float* __restrict__ bias,          // [280]
                      float* __restrict__ logits)              // [32768][280]
{
    __shared__ __align__(16) unsigned short sB[2][NPAD * LDB]; // 2 x 23040 B

    const int tid   = threadIdx.x;
    const int lane  = tid & 31;
    const int wave  = tid >> 5;
    const int half  = lane >> 4;       // 0: lanes 0-15, 1: lanes 16-31
    const int l15   = lane & 15;

    const int mBase = blockIdx.x * BM + wave * 16;
    const float* arow = hidden + (size_t)(mBase + l15) * IN_DIM;
    const int aKoff = half * 8;        // A frag: lanes>=16 hold K 8-15 / 24-31
    const int bKoff = half * 8;        // B frag: lanes>=16 hold K 8-15 / 24-31

    const uint32_t ldsBase0 = (uint32_t)(uintptr_t)&sB[0][0];
    const uint32_t ldsBase1 = (uint32_t)(uintptr_t)&sB[1][0];

    v8f acc[NT];
#pragma unroll
    for (int t = 0; t < NT; ++t)
#pragma unroll
        for (int r = 0; r < 8; ++r) acc[t][r] = 0.0f;

    // ---- prologue: DMA first B tile, load first A chunk (non-temporal) ----
    stageB_async(Wb, 0, ldsBase0, tid);
    const v4f* ap0 = (const v4f*)(arow + aKoff);
    v4f a0 = __builtin_nontemporal_load(ap0 + 0);   // K +0..3
    v4f a1 = __builtin_nontemporal_load(ap0 + 1);   // K +4..7
    v4f a2 = __builtin_nontemporal_load(ap0 + 4);   // K +16..19
    v4f a3 = __builtin_nontemporal_load(ap0 + 5);   // K +20..23
    wait_async0();
    __syncthreads();

    for (int kc = 0; kc < IN_DIM; kc += KC) {
        const int buf = (kc >> 5) & 1;
        const bool more = (kc + KC) < IN_DIM;

        // ---- overlap: DMA next B tile + load next A chunk while WMMAs run ----
        if (more) stageB_async(Wb, kc + KC, buf ? ldsBase0 : ldsBase1, tid);
        v4f n0 = a0, n1 = a1, n2 = a2, n3 = a3;
        if (more) {
            const v4f* ap = (const v4f*)(arow + (kc + KC) + aKoff);
            n0 = __builtin_nontemporal_load(ap + 0);
            n1 = __builtin_nontemporal_load(ap + 1);
            n2 = __builtin_nontemporal_load(ap + 4);
            n3 = __builtin_nontemporal_load(ap + 5);
        }

        // ---- convert current A chunk to bf16 fragment ----
        v16bf af;
        af[0]=(__bf16)a0[0];  af[1]=(__bf16)a0[1];  af[2]=(__bf16)a0[2];  af[3]=(__bf16)a0[3];
        af[4]=(__bf16)a1[0];  af[5]=(__bf16)a1[1];  af[6]=(__bf16)a1[2];  af[7]=(__bf16)a1[3];
        af[8]=(__bf16)a2[0];  af[9]=(__bf16)a2[1];  af[10]=(__bf16)a2[2]; af[11]=(__bf16)a2[3];
        af[12]=(__bf16)a3[0]; af[13]=(__bf16)a3[1]; af[14]=(__bf16)a3[2]; af[15]=(__bf16)a3[3];

        // ---- 18 WMMAs over the N tiles ----
        const unsigned short* sBc = &sB[buf][0];
#pragma unroll
        for (int t = 0; t < NT; ++t) {
            const v4f* bp = (const v4f*)(&sBc[(t * 16 + l15) * LDB + bKoff]);
            union { v4f f[2]; v16bf v; } u;
            u.f[0] = bp[0];
            u.f[1] = bp[2];
            acc[t] = __builtin_amdgcn_wmma_f32_16x16x32_bf16(
                false, af, false, u.v, (short)0, acc[t], false, false);
        }

        wait_async0();       // next tile's DMA done (overlapped with WMMAs above)
        __syncthreads();     // make it visible to all waves / protect read buffer
        a0 = n0; a1 = n1; a2 = n2; a3 = n3;
    }

    // ---- epilogue: bias + store (C/D layout: reg r -> M = r + 8*half, N = l15) ----
    const int mOff = half * 8;
#pragma unroll
    for (int t = 0; t < NT; ++t) {
        int n = t * 16 + l15;
        if (n < NCOLS) {
            float bv = bias[n];
            float* outp = logits + (size_t)(mBase + mOff) * NCOLS + n;
#pragma unroll
            for (int r = 0; r < 8; ++r)
                outp[(size_t)r * NCOLS] = acc[t][r] + bv;
        }
    }
}

// ---------------- Kernel 2: per-row cross-entropy + block partial sums ----------------
__global__ __launch_bounds__(256)
void ce_partial_kernel(const float* __restrict__ logits,
                       const int* __restrict__ labels,
                       float* __restrict__ partial)
{
    __shared__ float red[256];
    const int tid = threadIdx.x;
    const int r   = blockIdx.x * 256 + tid;      // exact: 3584*256 = 32768*28
    const float* row = logits + (size_t)r * 10;

    float x[10];
#pragma unroll
    for (int c = 0; c < 10; ++c) x[c] = row[c];

    float m = x[0];
#pragma unroll
    for (int c = 1; c < 10; ++c) m = fmaxf(m, x[c]);

    int lbl = labels[r];
    if ((unsigned)lbl > 9u) lbl = 0;

    float s = 0.0f, picked = 0.0f;
#pragma unroll
    for (int c = 0; c < 10; ++c) {
        float e = x[c] - m;
        s += __expf(e);
        picked += (c == lbl) ? e : 0.0f;
    }
    float nll = __logf(s) - picked;

    red[tid] = nll;
    __syncthreads();
#pragma unroll
    for (int st = 128; st > 0; st >>= 1) {
        if (tid < st) red[tid] += red[tid + st];
        __syncthreads();
    }
    if (tid == 0) partial[blockIdx.x] = red[0];
}

// ---------------- Kernel 3: final deterministic reduction ----------------
__global__ __launch_bounds__(256)
void ce_final_kernel(const float* __restrict__ partial, int nPartial,
                     float* __restrict__ loss)
{
    __shared__ float red[256];
    const int tid = threadIdx.x;
    float s = 0.0f;
    for (int i = tid; i < nPartial; i += 256) s += partial[i];
    red[tid] = s;
    __syncthreads();
#pragma unroll
    for (int st = 128; st > 0; st >>= 1) {
        if (tid < st) red[tid] += red[tid + st];
        __syncthreads();
    }
    if (tid == 0) loss[0] = red[0] * (1.0f / ((float)BATCH_SZ * 28.0f));
}

extern "C" void kernel_launch(void* const* d_in, const int* in_sizes, int n_in,
                              void* d_out, int out_size, void* d_ws, size_t ws_size,
                              hipStream_t stream)
{
    const float* hidden = (const float*)d_in[0];
    const int*   labels = (const int*)d_in[1];
    const float* W      = (const float*)d_in[2];
    const float* bias   = (const float*)d_in[3];
    float* out = (float*)d_out;                      // [0] = loss, [1..] = logits

    const int wElems = NCOLS * IN_DIM;               // 573440
    unsigned short* Wb = (unsigned short*)d_ws;      // 1,146,880 B
    float* partial = (float*)((char*)d_ws + (size_t)wElems * sizeof(unsigned short));

    // 1) W -> bf16
    wcvt_kernel<<<(wElems + 255) / 256, 256, 0, stream>>>(W, Wb, wElems);

    // 2) GEMM + bias -> logits
    gemm_bf16_kernel<<<BATCH_SZ / BM, 256, 0, stream>>>(hidden, Wb, bias, out + 1);

    // 3) cross-entropy
    const int rows = BATCH_SZ * 28;                  // 917504
    const int nblk = rows / 256;                     // 3584
    ce_partial_kernel<<<nblk, 256, 0, stream>>>(out + 1, labels, partial);
    ce_final_kernel<<<1, 256, 0, stream>>>(partial, nblk, out);
}
